// Adaptive1DLUTNet_69758858821773
// MI455X (gfx1250) — matrix operand order repaired and checked by reading, not compiled, not run on hardware
//
#include <hip/hip_runtime.h>
#include <hip/hip_bf16.h>
#include <math.h>

typedef __attribute__((ext_vector_type(16))) __bf16 v16bf;
typedef __attribute__((ext_vector_type(8)))  float  v8f;

// ---------- helpers ----------

__device__ __forceinline__ unsigned int f2bf(float f) {
  // round-to-nearest-even f32 -> bf16 bits (low 16 of result)
  unsigned int x = __float_as_uint(f);
  x += 0x7FFFu + ((x >> 16) & 1u);
  return (x >> 16) & 0xFFFFu;
}

__device__ __forceinline__ float bf2f(unsigned short u) {
  return __uint_as_float(((unsigned int)u) << 16);
}

__device__ __forceinline__ int reflect1(int i, int n) {
  // reflect padding, pad=1: -1 -> 1, n -> n-2
  if (i < 0) i = -i;
  if (i >= n) i = 2 * n - 2 - i;
  return i;
}

__device__ __forceinline__ float siluf(float x) {
  return x / (1.0f + __expf(-x));
}

// ---------- kernels ----------

// img (NCHW f32, 1024x1024, 3ch) -> x256 (NHWC bf16, 256x256, 32ch: 3 real + 29 zero pad)
__global__ void resize_nhwc_kernel(const float* __restrict__ img, unsigned short* __restrict__ out) {
  int idx = blockIdx.x * blockDim.x + threadIdx.x;   // one thread per output pixel
  const int total = 2 * 256 * 256;
  if (idx >= total) return;
  int p = idx & 65535;
  int b = idx >> 16;
  int oy = p >> 8, ox = p & 255;

  const float s = 1023.0f / 255.0f;
  float py = oy * s;
  int y0 = (int)py; if (y0 > 1022) y0 = 1022;
  float fy = py - (float)y0;
  float px = ox * s;
  int x0 = (int)px; if (x0 > 1022) x0 = 1022;
  float fx = px - (float)x0;

  unsigned int d[2] = {0u, 0u};
  #pragma unroll
  for (int c = 0; c < 3; ++c) {
    const float* pch = img + ((size_t)(b * 3 + c)) * 1024 * 1024;
    float v00 = pch[(size_t)y0 * 1024 + x0];
    float v01 = pch[(size_t)y0 * 1024 + x0 + 1];
    float v10 = pch[(size_t)(y0 + 1) * 1024 + x0];
    float v11 = pch[(size_t)(y0 + 1) * 1024 + x0 + 1];
    float a = v00 + fy * (v10 - v00);
    float bb = v01 + fy * (v11 - v01);
    float v = a + fx * (bb - a);
    d[c >> 1] |= f2bf(v) << ((c & 1) * 16);
  }
  uint4* o4 = (uint4*)(out + (size_t)idx * 32);
  o4[0] = make_uint4(d[0], d[1], 0u, 0u);
  o4[1] = make_uint4(0u, 0u, 0u, 0u);
  o4[2] = make_uint4(0u, 0u, 0u, 0u);
  o4[3] = make_uint4(0u, 0u, 0u, 0u);
}

// Pre-pack OIHW f32 weights into WMMA B-fragment order, bf16.
// K ordering: kk = (kh*3+kw)*Cinp + ci.  Layout: [(tile_n*chunks+chunk)*32 + lane]*8 + v dwords.
// dword v of lane: lo = B[k0+16*hf+2v][n], hi = B[k0+16*hf+2v+1][n]
__global__ void pack_weights_kernel(const float* __restrict__ w, unsigned int* __restrict__ wpack,
                                    int Cin_real, int Cinp, int Cout, int chunks) {
  int idx = blockIdx.x * blockDim.x + threadIdx.x;
  int total = (Cout / 16) * chunks * 256;
  if (idx >= total) return;
  int v      = idx & 7;
  int lane   = (idx >> 3) & 31;
  int t      = idx >> 8;
  int chunk  = t % chunks;
  int tile_n = t / chunks;
  int hf = lane >> 4;
  int n  = tile_n * 16 + (lane & 15);
  int kk = chunk * 32 + 16 * hf + 2 * v;
  int r  = kk / Cinp;          // filter tap 0..8
  int ci = kk - r * Cinp;      // even, ci+1 never wraps the tap
  unsigned int lo = 0, hi = 0;
  if (ci < Cin_real)     lo = f2bf(w[((size_t)n * Cin_real + ci) * 9 + r]);
  if (ci + 1 < Cin_real) hi = f2bf(w[((size_t)n * Cin_real + ci + 1) * 9 + r]);
  wpack[idx] = lo | (hi << 16);
}

// 3x3 conv (reflect pad 1) as implicit GEMM, WMMA bf16 -> f32, SiLU epilogue.
// Activations: bf16 NHWC (Cinp multiple of 32). Exact tiling: no predication.
// One wave = 16(M pixels) x 32(N channels): two WMMAs share one A fragment.
// Software-pipelined: next chunk's fragments are loaded before current WMMAs issue.
__global__ void conv3x3_wmma_kernel(const unsigned short* __restrict__ in,
                                    const uint4* __restrict__ wpack,
                                    const float* __restrict__ bias,
                                    unsigned short* __restrict__ out,
                                    int Cinp, int Cout, int Hin, int Win,
                                    int Hout, int Wout, int stride, int chunks) {
  const int WPB = 4;
  int wave = threadIdx.x >> 5;
  int lane = threadIdx.x & 31;
  int hf   = lane >> 4;
  int l16  = lane & 15;

  int m0 = (blockIdx.x * WPB + wave) * 16;
  int b  = blockIdx.z;

  const unsigned short* inb = in + (size_t)b * Hin * Win * Cinp;

  // A row owned by this lane
  int m  = m0 + l16;
  int oy = m / Wout;
  int ox = m - oy * Wout;

  // filter-tap state (uniform progression; per-lane spatial addresses)
  int kh = 0, kw = 0, ci_base = 0;
  int iy = reflect1(oy * stride - 1, Hin);
  int ix = reflect1(ox * stride - 1, Win);
  const unsigned short* arow = inb + (size_t)(iy * Win + ix) * Cinp;

  const uint4* bptr0 = wpack + ((size_t)(blockIdx.y * 2    ) * chunks * 32 + lane) * 2;
  const uint4* bptr1 = wpack + ((size_t)(blockIdx.y * 2 + 1) * chunks * 32 + lane) * 2;

  union Frag { v16bf v; uint4 q[2]; };
  Frag A, B0, B1;

  // prologue: load chunk 0 fragments
  {
    const unsigned short* ap = arow + 16 * hf;
    A.q[0]  = *(const uint4*)ap;
    A.q[1]  = *(const uint4*)(ap + 8);
    B0.q[0] = bptr0[0]; B0.q[1] = bptr0[1];
    B1.q[0] = bptr1[0]; B1.q[1] = bptr1[1];
  }

  v8f acc0 = {}, acc1 = {};
  for (int chunk = 0; chunk < chunks - 1; ++chunk) {
    // advance state to chunk+1 (uniform tap progression)
    ci_base += 32;
    if (ci_base == Cinp) {
      ci_base = 0;
      kw++; if (kw == 3) { kw = 0; kh++; }
      iy = reflect1(oy * stride - 1 + kh, Hin);
      ix = reflect1(ox * stride - 1 + kw, Win);
      arow = inb + (size_t)(iy * Win + ix) * Cinp;
    }
    bptr0 += 64;
    bptr1 += 64;

    // issue next chunk's loads before consuming current fragments
    Frag An, B0n, B1n;
    const unsigned short* ap = arow + ci_base + 16 * hf;
    An.q[0]  = *(const uint4*)ap;
    An.q[1]  = *(const uint4*)(ap + 8);
    B0n.q[0] = bptr0[0]; B0n.q[1] = bptr0[1];
    B1n.q[0] = bptr1[0]; B1n.q[1] = bptr1[1];

    acc0 = __builtin_amdgcn_wmma_f32_16x16x32_bf16(
        false, A.v, false, B0.v, (short)0, acc0, false, false);
    acc1 = __builtin_amdgcn_wmma_f32_16x16x32_bf16(
        false, A.v, false, B1.v, (short)0, acc1, false, false);

    A = An; B0 = B0n; B1 = B1n;
  }
  acc0 = __builtin_amdgcn_wmma_f32_16x16x32_bf16(
      false, A.v, false, B0.v, (short)0, acc0, false, false);
  acc1 = __builtin_amdgcn_wmma_f32_16x16x32_bf16(
      false, A.v, false, B1.v, (short)0, acc1, false, false);

  // D layout: vgpr r -> (M = m0 + 8*hf + r, N fixed per lane); bias + SiLU, bf16 NHWC store
  int n0 = blockIdx.y * 32 + l16;
  float bb0 = bias[n0];
  float bb1 = bias[n0 + 16];
  unsigned short* outb = out + (size_t)b * Hout * Wout * Cout;
  #pragma unroll
  for (int r = 0; r < 8; ++r) {
    int mm = m0 + hf * 8 + r;
    outb[(size_t)mm * Cout + n0]      = (unsigned short)f2bf(siluf(acc0[r] + bb0));
    outb[(size_t)mm * Cout + n0 + 16] = (unsigned short)f2bf(siluf(acc1[r] + bb1));
  }
}

// avg-pool c1..c4 (bf16 NHWC) to 8x8 and concat -> feat (bf16 NHWC, 480ch)
__global__ void pool_concat_kernel(const unsigned short* __restrict__ c1,
                                   const unsigned short* __restrict__ c2,
                                   const unsigned short* __restrict__ c3,
                                   const unsigned short* __restrict__ c4,
                                   unsigned short* __restrict__ feat) {
  int idx = blockIdx.x * blockDim.x + threadIdx.x;
  const int total = 2 * 480 * 64;
  if (idx >= total) return;
  int p  = idx % 64;
  int t  = idx / 64;
  int chout = t % 480;
  int b  = t / 480;
  int py = p >> 3, px = p & 7;

  const unsigned short* src; int k, C, Hs, ch = chout;
  if (ch < 32)       { src = c1; k = 16; C = 32;  Hs = 128; }
  else if (ch < 96)  { src = c2; k = 8;  C = 64;  Hs = 64;  ch -= 32;  }
  else if (ch < 224) { src = c3; k = 4;  C = 128; Hs = 32;  ch -= 96;  }
  else               { src = c4; k = 2;  C = 256; Hs = 16;  ch -= 224; }

  const unsigned short* sp = src + (size_t)b * Hs * Hs * C;
  float s = 0.0f;
  for (int yy = 0; yy < k; ++yy)
    for (int xx = 0; xx < k; ++xx)
      s += bf2f(sp[(size_t)((py * k + yy) * Hs + (px * k + xx)) * C + ch]);
  feat[((size_t)b * 64 + p) * 480 + chout] = (unsigned short)f2bf(s / (float)(k * k));
}

// 1x1 conv 128->27 + sigmoid: h1 (bf16 NHWC) -> wl (f32, [b][27][64])
__global__ void lin2_kernel(const unsigned short* __restrict__ h1, const float* __restrict__ w2,
                            const float* __restrict__ b2, float* __restrict__ wl) {
  int idx = blockIdx.x * blockDim.x + threadIdx.x;
  const int total = 2 * 27 * 64;
  if (idx >= total) return;
  int p = idx % 64;
  int t = idx / 64;
  int o = t % 27;
  int b = t / 27;
  const unsigned short* hp = h1 + (size_t)b * 64 * 128;
  float s = b2[o];
  for (int ci = 0; ci < 128; ++ci) s += bf2f(hp[(size_t)p * 128 + ci]) * w2[o * 128 + ci];
  wl[idx] = 1.0f / (1.0f + __expf(-s));
}

// Fused upsample(8x8 -> HxW) of predicted weights + 1D-LUT blend + clip.
// 4 pixels per thread, b128 in/out (HBM-bound pass: ~50MB total traffic).
__global__ void lut_apply_kernel(const float4* __restrict__ img4, const float* __restrict__ wl,
                                 const float* __restrict__ luts, float4* __restrict__ out4,
                                 int B, int H, int W) {
  int idx = blockIdx.x * blockDim.x + threadIdx.x;
  int W4 = W >> 2;
  int total = B * 3 * H * W4;
  if (idx >= total) return;
  int xq = idx % W4;
  int t  = idx / W4;
  int y  = t % H; t /= H;
  int c  = t % 3;
  int b  = t / 3;

  float4 p4 = img4[idx];
  float pin[4] = {p4.x, p4.y, p4.z, p4.w};

  float py = (float)y * (7.0f / (float)(H - 1));
  int y0 = (int)py; if (y0 > 6) y0 = 6;
  float fy = py - (float)y0;

  const float* wlb = wl + ((size_t)b * 27 + c * 9) * 64;
  const float* lc  = luts + (size_t)c * 9 * 256;

  float res[4];
  #pragma unroll
  for (int j = 0; j < 4; ++j) {
    int x = xq * 4 + j;
    float px = (float)x * (7.0f / (float)(W - 1));
    int x0 = (int)px; if (x0 > 6) x0 = 6;
    float fx = px - (float)x0;

    float pos = fminf(fmaxf(pin[j], 0.0f), 1.0f) * 255.0f;
    int i0 = (int)pos;
    if (i0 > 254) i0 = 254;
    float fr = pos - (float)i0;

    float acc = 0.0f;
    #pragma unroll
    for (int k = 0; k < 9; ++k) {
      const float* wp = wlb + k * 64 + y0 * 8 + x0;
      float w00 = wp[0], w01 = wp[1], w10 = wp[8], w11 = wp[9];
      float wa = w00 + fy * (w10 - w00);
      float wb = w01 + fy * (w11 - w01);
      float wv = wa + fx * (wb - wa);
      const float* lp = lc + k * 256;
      float v0 = lp[i0], v1 = lp[i0 + 1];
      acc += wv * (v0 + fr * (v1 - v0));
    }
    res[j] = fminf(fmaxf(acc, 0.0f), 1.0f);
  }
  out4[idx] = make_float4(res[0], res[1], res[2], res[3]);
}

// ---------- host launch ----------

extern "C" void kernel_launch(void* const* d_in, const int* in_sizes, int n_in,
                              void* d_out, int out_size, void* d_ws, size_t ws_size,
                              hipStream_t stream) {
  (void)in_sizes; (void)n_in; (void)out_size; (void)ws_size;
  const float* img  = (const float*)d_in[0];
  const float* c1w  = (const float*)d_in[1];
  const float* c1b  = (const float*)d_in[2];
  const float* c2w  = (const float*)d_in[3];
  const float* c2b  = (const float*)d_in[4];
  const float* c3w  = (const float*)d_in[5];
  const float* c3b  = (const float*)d_in[6];
  const float* c4w  = (const float*)d_in[7];
  const float* c4b  = (const float*)d_in[8];
  const float* l1w  = (const float*)d_in[9];
  const float* l1b  = (const float*)d_in[10];
  const float* l2w  = (const float*)d_in[11];
  const float* l2b  = (const float*)d_in[12];
  const float* luts = (const float*)d_in[13];
  float* out = (float*)d_out;

  // workspace carve (256B aligned chunks)
  char* ws = (char*)d_ws;
  size_t o = 0;
  auto carve = [&](size_t bytes) {
    char* p = ws + o;
    o += (bytes + 255) & ~(size_t)255;
    return p;
  };
  unsigned short* x256 = (unsigned short*)carve((size_t)2 * 65536 * 32 * 2);
  unsigned short* c1   = (unsigned short*)carve((size_t)2 * 16384 * 32 * 2);
  unsigned short* c2   = (unsigned short*)carve((size_t)2 * 4096 * 64 * 2);
  unsigned short* c3   = (unsigned short*)carve((size_t)2 * 1024 * 128 * 2);
  unsigned short* c4   = (unsigned short*)carve((size_t)2 * 256 * 256 * 2);
  unsigned short* feat = (unsigned short*)carve((size_t)2 * 64 * 480 * 2);
  unsigned short* h1   = (unsigned short*)carve((size_t)2 * 64 * 128 * 2);
  float*          wl   = (float*)carve((size_t)2 * 27 * 64 * 4);
  // packed weights: (Cout/16)*chunks*256 dwords each
  unsigned int* wp1 = (unsigned int*)carve((size_t)2  * 9   * 256 * 4);
  unsigned int* wp2 = (unsigned int*)carve((size_t)4  * 9   * 256 * 4);
  unsigned int* wp3 = (unsigned int*)carve((size_t)8  * 18  * 256 * 4);
  unsigned int* wp4 = (unsigned int*)carve((size_t)16 * 36  * 256 * 4);
  unsigned int* wp5 = (unsigned int*)carve((size_t)8  * 135 * 256 * 4);

  // weight packing (tiny, deterministic, every call)
  auto pack = [&](const float* w, unsigned int* wp, int CinR, int Cinp, int Cout, int chunks) {
    int total = (Cout / 16) * chunks * 256;
    pack_weights_kernel<<<(total + 255) / 256, 256, 0, stream>>>(w, wp, CinR, Cinp, Cout, chunks);
  };
  pack(c1w, wp1, 3,   32,  32,  9);
  pack(c2w, wp2, 32,  32,  64,  9);
  pack(c3w, wp3, 64,  64,  128, 18);
  pack(c4w, wp4, 128, 128, 256, 36);
  pack(l1w, wp5, 480, 480, 128, 135);

  // 1. img 1024 -> 256, NHWC bf16, channel-padded to 32
  {
    int total = 2 * 256 * 256;
    resize_nhwc_kernel<<<(total + 255) / 256, 256, 0, stream>>>(img, x256);
  }

  // 2. conv trunk (WMMA bf16, 16x32 tile per wave, pipelined)
  auto conv = [&](const unsigned short* in, const unsigned int* wp, const float* bias,
                  unsigned short* outp, int Cinp, int Cout, int Hin, int stride, int chunks) {
    int Hout = Hin / stride;
    int M = Hout * Hout;
    dim3 grid(M / 64, Cout / 32, 2);
    conv3x3_wmma_kernel<<<grid, 128, 0, stream>>>(in, (const uint4*)wp, bias, outp,
                                                  Cinp, Cout, Hin, Hin, Hout, Hout,
                                                  stride, chunks);
  };
  conv(x256, wp1, c1b, c1, 32,  32,  256, 2, 9);
  conv(c1,   wp2, c2b, c2, 32,  64,  128, 2, 9);
  conv(c2,   wp3, c3b, c3, 64,  128, 64,  2, 18);
  conv(c3,   wp4, c4b, c4, 128, 256, 32,  2, 36);

  // 3. pool + concat
  {
    int total = 2 * 480 * 64;
    pool_concat_kernel<<<(total + 255) / 256, 256, 0, stream>>>(c1, c2, c3, c4, feat);
  }

  // 4. lin1: 3x3 conv 480->128 stride 1 (WMMA)
  conv(feat, wp5, l1b, h1, 480, 128, 8, 1, 135);

  // 5. lin2: 1x1 conv 128->27 + sigmoid
  {
    int total = 2 * 27 * 64;
    lin2_kernel<<<(total + 255) / 256, 256, 0, stream>>>(h1, l2w, l2b, wl);
  }

  // 6. fused weight-upsample + LUT blend + clip (HBM-bound, float4 I/O)
  {
    int total = 2 * 3 * 1024 * 256;
    lut_apply_kernel<<<(total + 255) / 256, 256, 0, stream>>>(
        (const float4*)img, wl, luts, (float4*)out, 2, 1024, 1024);
  }
}
